// MoE_57758720196700
// MI455X (gfx1250) — compile-verified, matrix operand order checked by reading, not run on hardware
//
#include <hip/hip_runtime.h>
#include <cstdint>
#include <cstddef>

// ---------------- types ----------------
typedef __attribute__((ext_vector_type(16))) __bf16 v16bf;
typedef __attribute__((ext_vector_type(8)))  __bf16 v8bf;
typedef __attribute__((ext_vector_type(8)))  float  v8f;

typedef int v4i __attribute__((vector_size(16)));
typedef __attribute__((address_space(3))) void as3_void;
typedef __attribute__((address_space(1))) void as1_void;
typedef __attribute__((address_space(1))) v4i  as1_v4i;
typedef __attribute__((address_space(3))) v4i  as3_v4i;

// ---------------- problem sizes ----------------
#define Hdim  2048
#define Tdim  4096
#define Edim  8
#define Cdim  1536
#define Idim  1024
#define ISdim 1024
#define APAD  40     // LDS row stride in bf16 elems (16B aligned, bank-spread)

// -------- CDNA5 async global->LDS copy (ASYNCcnt path), with fallback ----
#if defined(__has_builtin)
#if __has_builtin(__builtin_amdgcn_global_load_async_to_lds_b128)
#define HAVE_ASYNC_COPY 1
#endif
#if __has_builtin(__builtin_amdgcn_s_wait_asynccnt)
#define HAVE_WAIT_ASYNC 1
#endif
#endif

static __device__ inline void async_copy16(const void* g, void* l) {
#if defined(HAVE_ASYNC_COPY)
  __builtin_amdgcn_global_load_async_to_lds_b128(
      (as1_v4i*)(as1_void*)(void*)(uintptr_t)g,
      (as3_v4i*)(as3_void*)l, 0, 0);
#else
  *(v8bf*)l = *(const v8bf*)g;
#endif
}
static __device__ inline void wait_async0() {
#if defined(HAVE_ASYNC_COPY)
#if defined(HAVE_WAIT_ASYNC)
  __builtin_amdgcn_s_wait_asynccnt(0);
#else
  asm volatile("s_wait_asynccnt 0x0" ::: "memory");
#endif
#endif
}

// Load one 16x32 bf16 WMMA operand fragment from an LDS tile stored
// row-major with APAD stride. Per ISA 7.12.2 (16-bit A/B layout), lane l
// (m = l&15, kh = l>>4) needs elements [kh*8 .. +7] and [16+kh*8 .. +7]
// of its row: exactly two ds_load_b128.
static __device__ inline v16bf lds_frag(const __bf16* base, int row, int kh) {
  const __bf16* p = base + row * APAD + kh * 8;
  v8bf lo = *(const v8bf*)(p);
  v8bf hi = *(const v8bf*)(p + 16);
  return __builtin_shufflevector(lo, hi, 0,1,2,3,4,5,6,7,8,9,10,11,12,13,14,15);
}

// =====================================================================
// Kernel 1: RMSNorm + gate (softmax, top-2, renorm) + dispatch scatter
// One block (256 thr) per token.
// =====================================================================
__global__ __launch_bounds__(256) void k_gate_dispatch(
    const float* __restrict__ hs, const float* __restrict__ rmsw,
    const float* __restrict__ gw,
    __bf16* __restrict__ bufE, int* __restrict__ tk_e, int* __restrict__ tk_p,
    float* __restrict__ tk_w, int* __restrict__ counts)
{
  const int t    = blockIdx.x;
  const int tid  = threadIdx.x;
  const int lane = tid & 31;
  const int wave = tid >> 5;

  __shared__ float red[8];
  __shared__ float red8[8][8];
  __shared__ float slog[8];
  __shared__ float sRms;
  __shared__ int   sIdx[4];

  const int h0 = tid * 8;
  const float4 fa = *(const float4*)(hs + (size_t)t * Hdim + h0);
  const float4 fb = *(const float4*)(hs + (size_t)t * Hdim + h0 + 4);
  float x[8] = {fa.x, fa.y, fa.z, fa.w, fb.x, fb.y, fb.z, fb.w};

  float ss = 0.f;
  #pragma unroll
  for (int i = 0; i < 8; ++i) ss += x[i] * x[i];
  #pragma unroll
  for (int off = 16; off > 0; off >>= 1) ss += __shfl_xor(ss, off, 32);
  if (lane == 0) red[wave] = ss;
  __syncthreads();
  if (tid == 0) {
    float tot = 0.f;
    for (int w = 0; w < 8; ++w) tot += red[w];
    sRms = rsqrtf(tot / (float)Hdim + 1e-6f);
  }
  __syncthreads();
  const float rms = sRms;

  float  xn[8];
  __bf16 xb[8];
  #pragma unroll
  for (int i = 0; i < 8; ++i) {
    xn[i] = x[i] * rms * rmsw[h0 + i];
    xb[i] = (__bf16)xn[i];
  }

  float lg[8] = {0.f, 0.f, 0.f, 0.f, 0.f, 0.f, 0.f, 0.f};
  #pragma unroll
  for (int e = 0; e < 8; ++e) {
    const float* ge = gw + (size_t)e * Hdim + h0;
    float s = 0.f;
    #pragma unroll
    for (int i = 0; i < 8; ++i) s += xn[i] * ge[i];
    lg[e] = s;
  }
  #pragma unroll
  for (int e = 0; e < 8; ++e) {
    #pragma unroll
    for (int off = 16; off > 0; off >>= 1) lg[e] += __shfl_xor(lg[e], off, 32);
  }
  if (lane == 0) {
    #pragma unroll
    for (int e = 0; e < 8; ++e) red8[wave][e] = lg[e];
  }
  __syncthreads();
  if (tid < 8) {                        // deterministic cross-wave reduce
    float s = 0.f;
    for (int w = 0; w < 8; ++w) s += red8[w][tid];
    slog[tid] = s;
  }
  __syncthreads();

  if (tid == 0) {
    float mx = slog[0];
    for (int e = 1; e < 8; ++e) mx = fmaxf(mx, slog[e]);
    float p[8], sum = 0.f;
    for (int e = 0; e < 8; ++e) { p[e] = expf(slog[e] - mx); sum += p[e]; }
    for (int e = 0; e < 8; ++e) p[e] /= sum;
    int e0 = 0;
    for (int e = 1; e < 8; ++e) if (p[e] > p[e0]) e0 = e;
    int e1 = (e0 == 0) ? 1 : 0;
    for (int e = 0; e < 8; ++e) if (e != e0 && p[e] > p[e1]) e1 = e;
    float w0 = p[e0], w1 = p[e1], wn = w0 + w1;
    w0 /= wn; w1 /= wn;
    int p0 = atomicAdd(&counts[e0], 1);
    int p1 = atomicAdd(&counts[e1], 1);
    tk_e[t * 2] = e0; tk_e[t * 2 + 1] = e1;
    tk_p[t * 2] = p0; tk_p[t * 2 + 1] = p1;
    tk_w[t * 2] = w0; tk_w[t * 2 + 1] = w1;
    sIdx[0] = e0; sIdx[1] = p0; sIdx[2] = e1; sIdx[3] = p1;
  }
  __syncthreads();

  v8bf v;
  #pragma unroll
  for (int i = 0; i < 8; ++i) v[i] = xb[i];
  const int e0 = sIdx[0], p0 = sIdx[1], e1 = sIdx[2], p1 = sIdx[3];
  if (p0 < Cdim) *(v8bf*)(bufE + ((size_t)e0 * Cdim + p0) * Hdim + h0) = v;
  if (p1 < Cdim) *(v8bf*)(bufE + ((size_t)e1 * Cdim + p1) * Hdim + h0) = v;
}

// =====================================================================
// Kernel 2: paired GEMM + fused silu(g)*u epilogue (bf16 out).
// Block tile: 128 rows x 64 cols (x2 halves). 8 waves as 4x2, each wave
// owns 32x32(x2 halves): 2 A-frags + 4 B-frags -> 8 WMMA per K-step.
// A tile: async global->LDS; B tile: fp32 fetch staged in regs, convert
// + transposed ds_store after the WMMA burst.
// =====================================================================
__global__ __launch_bounds__(256) void k_gemm_silu_pair(
    const __bf16* __restrict__ A, long long sAz, int lda,
    const float* __restrict__ B1, const float* __restrict__ B2,
    long long sBz, int ldb,
    __bf16* __restrict__ O, long long sOz, int ldo, int Kd)
{
  __shared__ __bf16 sA[2][128 * APAD];
  __shared__ __bf16 sB[2][128 * APAD];   // [n][k]; rows 0..63 = B1, 64..127 = B2

  const int tid  = threadIdx.x;
  const int lane = tid & 31;
  const int wave = tid >> 5;
  const int m16  = lane & 15;
  const int kh   = lane >> 4;
  const int mw   = wave & 3;     // row group: 32 rows
  const int nw   = wave >> 2;    // col group: 32 cols per half
  const int z        = blockIdx.z;
  const int rowBase  = blockIdx.y * 128;
  const int colBase  = blockIdx.x * 64;

  const __bf16* Az  = A  + (size_t)z * sAz;
  const float*  B1z = B1 + (size_t)z * sBz;
  const float*  B2z = B2 + (size_t)z * sBz;
  __bf16*       Oz  = O  + (size_t)z * sOz;

  v8f acc[2][4];
  #pragma unroll
  for (int mi = 0; mi < 2; ++mi)
    #pragma unroll
    for (int j = 0; j < 4; ++j)
      #pragma unroll
      for (int i = 0; i < 8; ++i) acc[mi][j][i] = 0.f;

  auto issueA = [&](int bi, int kk) {
    #pragma unroll
    for (int r = 0; r < 2; ++r) {
      int c = tid * 2 + r;
      int ar = c >> 2, co = (c & 3) * 8;
      async_copy16(Az + (size_t)(rowBase + ar) * lda + kk + co,
                   &sA[bi][ar * APAD + co]);
    }
  };
  auto fetchB = [&](int kk, float4* st) {
    #pragma unroll
    for (int r = 0; r < 4; ++r) {
      int c    = tid + r * 256;
      int half = c >> 9, cc = c & 511;
      int kr   = cc >> 4, n4 = (cc & 15) * 4;
      st[r] = *(const float4*)((half ? B2z : B1z) +
                               (size_t)(kk + kr) * ldb + colBase + n4);
    }
  };
  auto storeB = [&](int bi, const float4* st) {
    #pragma unroll
    for (int r = 0; r < 4; ++r) {
      int c    = tid + r * 256;
      int half = c >> 9, cc = c & 511;
      int kr   = cc >> 4, n4 = (cc & 15) * 4;
      __bf16* d = &sB[bi][(half * 64 + n4) * APAD + kr];
      d[0 * APAD] = (__bf16)st[r].x; d[1 * APAD] = (__bf16)st[r].y;
      d[2 * APAD] = (__bf16)st[r].z; d[3 * APAD] = (__bf16)st[r].w;
    }
  };

  const int nk = Kd >> 5;
  {
    float4 st[4];
    issueA(0, 0);
    fetchB(0, st);
    storeB(0, st);
    wait_async0();
  }
  __syncthreads();

  for (int k = 0; k < nk; ++k) {
    const int cur = k & 1;
    const bool pf = (k + 1 < nk);
    float4 st[4];
    if (pf) { issueA(cur ^ 1, (k + 1) * 32); fetchB((k + 1) * 32, st); }

    // all fragments for this K-step -> one dscnt wait, then WMMA burst
    v16bf a0 = lds_frag(sA[cur], mw * 32 + m16, kh);
    v16bf a1 = lds_frag(sA[cur], mw * 32 + 16 + m16, kh);
    v16bf bf0 = lds_frag(sB[cur],      nw * 32 +      m16, kh);
    v16bf bf1 = lds_frag(sB[cur],      nw * 32 + 16 + m16, kh);
    v16bf bf2 = lds_frag(sB[cur], 64 + nw * 32 +      m16, kh);
    v16bf bf3 = lds_frag(sB[cur], 64 + nw * 32 + 16 + m16, kh);

    acc[0][0] = __builtin_amdgcn_wmma_f32_16x16x32_bf16(false, a0, false, bf0, (short)0, acc[0][0], false, false);
    acc[0][1] = __builtin_amdgcn_wmma_f32_16x16x32_bf16(false, a0, false, bf1, (short)0, acc[0][1], false, false);
    acc[0][2] = __builtin_amdgcn_wmma_f32_16x16x32_bf16(false, a0, false, bf2, (short)0, acc[0][2], false, false);
    acc[0][3] = __builtin_amdgcn_wmma_f32_16x16x32_bf16(false, a0, false, bf3, (short)0, acc[0][3], false, false);
    acc[1][0] = __builtin_amdgcn_wmma_f32_16x16x32_bf16(false, a1, false, bf0, (short)0, acc[1][0], false, false);
    acc[1][1] = __builtin_amdgcn_wmma_f32_16x16x32_bf16(false, a1, false, bf1, (short)0, acc[1][1], false, false);
    acc[1][2] = __builtin_amdgcn_wmma_f32_16x16x32_bf16(false, a1, false, bf2, (short)0, acc[1][2], false, false);
    acc[1][3] = __builtin_amdgcn_wmma_f32_16x16x32_bf16(false, a1, false, bf3, (short)0, acc[1][3], false, false);

    if (pf) storeB(cur ^ 1, st);   // loadcnt wait lands after the WMMA burst
    wait_async0();
    __syncthreads();
  }

  // epilogue: silu(g) * u -> bf16
  #pragma unroll
  for (int mi = 0; mi < 2; ++mi)
    #pragma unroll
    for (int j = 0; j < 2; ++j)
      #pragma unroll
      for (int i = 0; i < 8; ++i) {
        float g = acc[mi][j][i], u = acc[mi][j + 2][i];
        float s = g / (1.f + expf(-g));
        int m = rowBase + mw * 32 + mi * 16 + kh * 8 + i;
        int n = colBase + nw * 32 + j * 16 + m16;
        Oz[(size_t)m * ldo + n] = (__bf16)(s * u);
      }
}

// =====================================================================
// Kernel 3: plain GEMM, 128x128 block tile, 4x2 wave grid (32x64/wave).
//  mode 0: Obf = bf16(A@B)
//  mode 1: Of  = r1 + r2 + (A@B)   (fp32, final output)
// =====================================================================
__global__ __launch_bounds__(256) void k_gemm_plain(
    const __bf16* __restrict__ A, long long sAz, int lda,
    const float* __restrict__ B, long long sBz, int ldb,
    __bf16* __restrict__ Obf, float* __restrict__ Of, long long sOz, int ldo,
    int Kd, int mode,
    const float* __restrict__ r1, const float* __restrict__ r2)
{
  __shared__ __bf16 sA[2][128 * APAD];
  __shared__ __bf16 sB[2][128 * APAD];

  const int tid  = threadIdx.x;
  const int lane = tid & 31;
  const int wave = tid >> 5;
  const int m16  = lane & 15;
  const int kh   = lane >> 4;
  const int mw   = wave & 3;     // 32-row group
  const int nw   = wave >> 2;    // 64-col group
  const int z       = blockIdx.z;
  const int rowBase = blockIdx.y * 128;
  const int colBase = blockIdx.x * 128;

  const __bf16* Az = A + (size_t)z * sAz;
  const float*  Bz = B + (size_t)z * sBz;

  v8f acc[2][4];
  #pragma unroll
  for (int mi = 0; mi < 2; ++mi)
    #pragma unroll
    for (int j = 0; j < 4; ++j)
      #pragma unroll
      for (int i = 0; i < 8; ++i) acc[mi][j][i] = 0.f;

  auto issueA = [&](int bi, int kk) {
    #pragma unroll
    for (int r = 0; r < 2; ++r) {
      int c = tid * 2 + r;
      int ar = c >> 2, co = (c & 3) * 8;
      async_copy16(Az + (size_t)(rowBase + ar) * lda + kk + co,
                   &sA[bi][ar * APAD + co]);
    }
  };
  auto fetchB = [&](int kk, float4* st) {
    #pragma unroll
    for (int r = 0; r < 4; ++r) {
      int c  = tid + r * 256;
      int kr = c >> 5, n4 = (c & 31) * 4;
      st[r] = *(const float4*)(Bz + (size_t)(kk + kr) * ldb + colBase + n4);
    }
  };
  auto storeB = [&](int bi, const float4* st) {
    #pragma unroll
    for (int r = 0; r < 4; ++r) {
      int c  = tid + r * 256;
      int kr = c >> 5, n4 = (c & 31) * 4;
      __bf16* d = &sB[bi][n4 * APAD + kr];
      d[0 * APAD] = (__bf16)st[r].x; d[1 * APAD] = (__bf16)st[r].y;
      d[2 * APAD] = (__bf16)st[r].z; d[3 * APAD] = (__bf16)st[r].w;
    }
  };

  const int nk = Kd >> 5;
  {
    float4 st[4];
    issueA(0, 0);
    fetchB(0, st);
    storeB(0, st);
    wait_async0();
  }
  __syncthreads();

  for (int k = 0; k < nk; ++k) {
    const int cur = k & 1;
    const bool pf = (k + 1 < nk);
    float4 st[4];
    if (pf) { issueA(cur ^ 1, (k + 1) * 32); fetchB((k + 1) * 32, st); }

    v16bf a0 = lds_frag(sA[cur], mw * 32 + m16, kh);
    v16bf a1 = lds_frag(sA[cur], mw * 32 + 16 + m16, kh);
    v16bf bf0 = lds_frag(sB[cur], nw * 64 +      m16, kh);
    v16bf bf1 = lds_frag(sB[cur], nw * 64 + 16 + m16, kh);
    v16bf bf2 = lds_frag(sB[cur], nw * 64 + 32 + m16, kh);
    v16bf bf3 = lds_frag(sB[cur], nw * 64 + 48 + m16, kh);

    acc[0][0] = __builtin_amdgcn_wmma_f32_16x16x32_bf16(false, a0, false, bf0, (short)0, acc[0][0], false, false);
    acc[0][1] = __builtin_amdgcn_wmma_f32_16x16x32_bf16(false, a0, false, bf1, (short)0, acc[0][1], false, false);
    acc[0][2] = __builtin_amdgcn_wmma_f32_16x16x32_bf16(false, a0, false, bf2, (short)0, acc[0][2], false, false);
    acc[0][3] = __builtin_amdgcn_wmma_f32_16x16x32_bf16(false, a0, false, bf3, (short)0, acc[0][3], false, false);
    acc[1][0] = __builtin_amdgcn_wmma_f32_16x16x32_bf16(false, a1, false, bf0, (short)0, acc[1][0], false, false);
    acc[1][1] = __builtin_amdgcn_wmma_f32_16x16x32_bf16(false, a1, false, bf1, (short)0, acc[1][1], false, false);
    acc[1][2] = __builtin_amdgcn_wmma_f32_16x16x32_bf16(false, a1, false, bf2, (short)0, acc[1][2], false, false);
    acc[1][3] = __builtin_amdgcn_wmma_f32_16x16x32_bf16(false, a1, false, bf3, (short)0, acc[1][3], false, false);

    if (pf) storeB(cur ^ 1, st);
    wait_async0();
    __syncthreads();
  }

  if (mode == 0) {
    __bf16* Oz = Obf + (size_t)z * sOz;
    #pragma unroll
    for (int mi = 0; mi < 2; ++mi)
      #pragma unroll
      for (int j = 0; j < 4; ++j)
        #pragma unroll
        for (int i = 0; i < 8; ++i) {
          int m = rowBase + mw * 32 + mi * 16 + kh * 8 + i;
          int n = colBase + nw * 64 + j * 16 + m16;
          Oz[(size_t)m * ldo + n] = (__bf16)acc[mi][j][i];
        }
  } else {
    float* Oz = Of + (size_t)z * sOz;
    #pragma unroll
    for (int mi = 0; mi < 2; ++mi)
      #pragma unroll
      for (int j = 0; j < 4; ++j)
        #pragma unroll
        for (int i = 0; i < 8; ++i) {
          int m = rowBase + mw * 32 + mi * 16 + kh * 8 + i;
          int n = colBase + nw * 64 + j * 16 + m16;
          size_t idx = (size_t)m * ldo + n;
          Oz[idx] = r1[idx] + r2[idx] + acc[mi][j][i];
        }
  }
}

// =====================================================================
// Kernel 4: combine — gather two expert rows per token, weight, sum.
// =====================================================================
__global__ __launch_bounds__(256) void k_combine(
    const __bf16* __restrict__ y, const int* __restrict__ tk_e,
    const int* __restrict__ tk_p, const float* __restrict__ tk_w,
    float* __restrict__ routedF, __bf16* __restrict__ routedB)
{
  const int t = blockIdx.x, tid = threadIdx.x;
  const int e0 = tk_e[t * 2], e1 = tk_e[t * 2 + 1];
  const int p0 = tk_p[t * 2], p1 = tk_p[t * 2 + 1];
  const float w0 = tk_w[t * 2], w1 = tk_w[t * 2 + 1];
  const int h0 = tid * 8;

  float r[8] = {0.f, 0.f, 0.f, 0.f, 0.f, 0.f, 0.f, 0.f};
  if (p0 < Cdim) {
    v8bf v = *(const v8bf*)(y + ((size_t)e0 * Cdim + p0) * Hdim + h0);
    #pragma unroll
    for (int i = 0; i < 8; ++i) r[i] += w0 * (float)v[i];
  }
  if (p1 < Cdim) {
    v8bf v = *(const v8bf*)(y + ((size_t)e1 * Cdim + p1) * Hdim + h0);
    #pragma unroll
    for (int i = 0; i < 8; ++i) r[i] += w1 * (float)v[i];
  }
  float4 o0 = {r[0], r[1], r[2], r[3]};
  float4 o1 = {r[4], r[5], r[6], r[7]};
  *(float4*)(routedF + (size_t)t * Hdim + h0)     = o0;
  *(float4*)(routedF + (size_t)t * Hdim + h0 + 4) = o1;
  v8bf rb;
  #pragma unroll
  for (int i = 0; i < 8; ++i) rb[i] = (__bf16)r[i];
  *(v8bf*)(routedB + (size_t)t * Hdim + h0) = rb;
}

// =====================================================================
// Host launcher
// =====================================================================
extern "C" void kernel_launch(void* const* d_in, const int* in_sizes, int n_in,
                              void* d_out, int out_size, void* d_ws, size_t ws_size,
                              hipStream_t stream) {
  (void)in_sizes; (void)n_in; (void)out_size; (void)ws_size;
  const float* hs      = (const float*)d_in[0];
  const float* rms_w   = (const float*)d_in[1];
  const float* gate_w  = (const float*)d_in[2];
  const float* w1w3    = (const float*)d_in[3];
  const float* w2      = (const float*)d_in[4];
  const float* sh_gate = (const float*)d_in[5];
  const float* sh_up   = (const float*)d_in[6];
  const float* sh_down = (const float*)d_in[7];
  float* out = (float*)d_out;

  char* ws = (char*)d_ws;
  size_t off = 0;
  auto carve = [&](size_t bytes) -> char* {
    char* p = ws + off;
    off = (off + bytes + 255) & ~(size_t)255;
    return p;
  };
  __bf16* bufE    = (__bf16*)carve((size_t)Edim * Cdim * Hdim * 2);
  __bf16* act     = (__bf16*)carve((size_t)Edim * Cdim * Idim * 2);
  __bf16* yb      = (__bf16*)carve((size_t)Edim * Cdim * Hdim * 2);
  float*  routedF = (float*) carve((size_t)Tdim * Hdim * 4);
  __bf16* routedB = (__bf16*)carve((size_t)Tdim * Hdim * 2);
  __bf16* actsh   = (__bf16*)carve((size_t)Tdim * ISdim * 2);
  int*    tk_e    = (int*)   carve((size_t)Tdim * 2 * 4);
  int*    tk_p    = (int*)   carve((size_t)Tdim * 2 * 4);
  float*  tk_w    = (float*) carve((size_t)Tdim * 2 * 4);
  int*    counts  = (int*)   carve((size_t)Edim * 4);

  (void)hipMemsetAsync(counts, 0, sizeof(int) * Edim, stream);

  // 1) RMSNorm + gate + dispatch
  k_gate_dispatch<<<Tdim, 256, 0, stream>>>(hs, rms_w, gate_w, bufE,
                                            tk_e, tk_p, tk_w, counts);

  // 2) expert up-projection, fused silu*up: [E] (C x H) @ (H x 2I) -> (C x I)
  k_gemm_silu_pair<<<dim3(Idim / 64, Cdim / 128, Edim), 256, 0, stream>>>(
      bufE, (long long)Cdim * Hdim, Hdim,
      w1w3, w1w3 + Idim, (long long)Hdim * 2 * Idim, 2 * Idim,
      act, (long long)Cdim * Idim, Idim, Hdim);

  // 3) expert down-projection: [E] (C x I) @ (I x H) -> (C x H) bf16
  k_gemm_plain<<<dim3(Hdim / 128, Cdim / 128, Edim), 256, 0, stream>>>(
      act, (long long)Cdim * Idim, Idim,
      w2, (long long)Idim * Hdim, Hdim,
      yb, nullptr, (long long)Cdim * Hdim, Hdim,
      Idim, 0, nullptr, nullptr);

  // 4) combine back per token
  k_combine<<<Tdim, 256, 0, stream>>>(yb, tk_e, tk_p, tk_w, routedF, routedB);

  // 5) shared expert up: (T x H) @ (H x IS) pair -> (T x IS)
  k_gemm_silu_pair<<<dim3(ISdim / 64, Tdim / 128, 1), 256, 0, stream>>>(
      routedB, 0, Hdim,
      sh_gate, sh_up, 0, ISdim,
      actsh, 0, ISdim, Hdim);

  // 6) shared expert down + residual + routed -> fp32 output
  k_gemm_plain<<<dim3(Hdim / 128, Tdim / 128, 1), 256, 0, stream>>>(
      actsh, 0, ISdim,
      sh_down, 0, Hdim,
      nullptr, out, 0, Hdim,
      ISdim, 1, hs, routedF);
}
